// SimpleMLPDecoder_84920093376638
// MI455X (gfx1250) — matrix operand order repaired and checked by reading, compiled, not run on hardware
//
#include <hip/hip_runtime.h>
#include <hip/hip_bf16.h>
#include <math.h>

// ---------------- types ----------------
typedef __attribute__((ext_vector_type(16))) __bf16       v16bf;
typedef __attribute__((ext_vector_type(8)))  float        v8f;
typedef __attribute__((ext_vector_type(8)))  unsigned int v8u;
// TDM descriptor groups (probe-documented builtin signature)
typedef __attribute__((ext_vector_type(4))) unsigned int tdm_g0_t;
typedef __attribute__((ext_vector_type(8))) int          tdm_g1_t;
typedef __attribute__((ext_vector_type(4))) int          tdm_g2_t;

// f32 -> bf16, round-to-nearest-even (bit-twiddle: always compiles)
static __device__ __forceinline__ unsigned short bfbits(float f) {
  unsigned u = __builtin_bit_cast(unsigned, f);
  u += 0x7FFFu + ((u >> 16) & 1u);
  return (unsigned short)(u >> 16);
}

static __device__ __forceinline__ float gelu_exact(float x) {
  return 0.5f * x * (1.0f + erff(x * 0.70710678118654752f));
}

// butterfly sum across a 16-lane half (xor masks 1,2,4,8 stay in the half)
static __device__ __forceinline__ float hredux16(float v) {
  v += __shfl_xor(v, 1, 32);
  v += __shfl_xor(v, 2, 32);
  v += __shfl_xor(v, 4, 32);
  v += __shfl_xor(v, 8, 32);
  return v;
}

#define D_IN  256
#define H1    128
#define H2    64
#define WAVES 8
#define LDS_W1T   (H1 * D_IN)           // 32768 bf16 elems (64 KB)
#define LDS_W2T   (H2 * H1)             // 8192  bf16 elems (16 KB)
#define LDS_H1S   (WAVES * 16 * H1)     // 16384 bf16 elems (32 KB)
#define SMEM_BYTES ((LDS_W1T + LDS_W2T + LDS_H1S) * 2)  // 114688 B
#define WPACK_ELEMS (LDS_W1T + LDS_W2T)                 // 40960 (81920 B)

// =====================================================================
// Prep kernel 1: fp32 feature table -> bf16 table in workspace
// =====================================================================
__global__ void feat_to_bf16(const float* __restrict__ in,
                             unsigned short* __restrict__ outp, int n4) {
  int i = blockIdx.x * blockDim.x + threadIdx.x;
  if (i < n4) {
    float4 v = ((const float4*)in)[i];
    uint2 t;
    t.x = (unsigned)bfbits(v.x) | ((unsigned)bfbits(v.y) << 16);
    t.y = (unsigned)bfbits(v.z) | ((unsigned)bfbits(v.w) << 16);
    ((uint2*)outp)[i] = t;
  }
}

// =====================================================================
// Prep kernel 2: pack W1^T [128][256] bf16 then W2^T [64][128] bf16
// =====================================================================
__global__ __launch_bounds__(256, 1)
void weights_pack(const float* __restrict__ W1, const float* __restrict__ W2,
                  unsigned short* __restrict__ wpack) {
  const int k = threadIdx.x;                    // 0..255 -> k-row of W1
  {
    const float4* wrow = (const float4*)(W1 + k * H1);
    for (int n4 = 0; n4 < H1 / 4; ++n4) {
      float4 v = wrow[n4];
      int n = n4 * 4;
      wpack[(n + 0) * D_IN + k] = bfbits(v.x);
      wpack[(n + 1) * D_IN + k] = bfbits(v.y);
      wpack[(n + 2) * D_IN + k] = bfbits(v.z);
      wpack[(n + 3) * D_IN + k] = bfbits(v.w);
    }
  }
  if (k < H1) {
    const float4* wrow = (const float4*)(W2 + k * H2);
    for (int n4 = 0; n4 < H2 / 4; ++n4) {
      float4 v = wrow[n4];
      int n = n4 * 4;
      wpack[LDS_W1T + (n + 0) * H1 + k] = bfbits(v.x);
      wpack[LDS_W1T + (n + 1) * H1 + k] = bfbits(v.y);
      wpack[LDS_W1T + (n + 2) * H1 + k] = bfbits(v.z);
      wpack[LDS_W1T + (n + 3) * H1 + k] = bfbits(v.w);
    }
  }
}

// =====================================================================
// FAST main kernel: bf16 features + TDM-staged bf16 weights
// =====================================================================
__global__ __launch_bounds__(256, 1)
void mlp_edge_decoder_fast(
    const unsigned short* __restrict__ fdrug,
    const unsigned short* __restrict__ fdis,
    const unsigned short* __restrict__ wpack,
    const int*   __restrict__ src,  const int*   __restrict__ dst,
    const float* __restrict__ b1, const float* __restrict__ g1,
    const float* __restrict__ be1,
    const float* __restrict__ b2, const float* __restrict__ g2,
    const float* __restrict__ be2,
    const float* __restrict__ W3, const float* __restrict__ b3p,
    float* __restrict__ out, int E)
{
  extern __shared__ __align__(128) unsigned short smem[];
  unsigned short* w1t = smem;                       // [n:128][k:256] bf16
  unsigned short* w2t = smem + LDS_W1T;             // [n:64][k:128]  bf16
  unsigned short* h1s = smem + LDS_W1T + LDS_W2T;   // per-wave [16][128] bf16

  const int tid  = threadIdx.x;
  const int wave = tid >> 5;

  // ---- stage packed bf16 weights (80KB) into LDS via Tensor Data Mover ----
#if __has_builtin(__builtin_amdgcn_tensor_load_to_lds)
  if (wave == 0) {
    unsigned lds_base = (unsigned)(size_t)smem;           // LDS byte offset
    unsigned long long ga = (unsigned long long)(size_t)wpack;
    // D# group0: count=1, lds_addr, global_addr, type=2 ("image")
    tdm_g0_t g0 = { 1u, lds_base, (unsigned)ga,
                    (unsigned)((ga >> 32) & 0x1FFFFFFu) | (2u << 30) };
    // D# group1: data_size=8B, tensor_dim0=tile_dim0=10240 (80KB), dim1=1,
    //            tile_dim1=1, tensor_dim0_stride=10240
    tdm_g1_t g1d = { (int)0x00030000, (int)0x28000000, (int)0x00010000,
                     (int)0x28000000, 1, 10240, 0, 0 };
    tdm_g2_t g2d = { 1, 1, 0, 0 };   // tensor_dim2/3 = 1, tile_dim3 unused
    tdm_g2_t g3d = { 0, 0, 0, 0 };
#if defined(__clang_major__) && __clang_major__ >= 23
    tdm_g1_t z8 = { 0, 0, 0, 0, 0, 0, 0, 0 };
    __builtin_amdgcn_tensor_load_to_lds(g0, g1d, g2d, g3d, z8, 0);
#else
    __builtin_amdgcn_tensor_load_to_lds(g0, g1d, g2d, g3d, 0);
#endif
    __builtin_amdgcn_s_wait_tensorcnt(0);
  }
#else
  for (int i = tid; i < WPACK_ELEMS; i += 256) smem[i] = wpack[i];
#endif
  __syncthreads();

  const int lane = tid & 31;
  const int l    = lane & 15;   // column / row-within-tile index
  const int hi   = lane >> 4;   // which K/M half of the fragment
  unsigned short* h1w = h1s + wave * (16 * H1);

  const int n_tiles = (E + 15) >> 4;
  const int tile = blockIdx.x * WAVES + wave;   // wave-uniform
  if (tile < n_tiles) {
    int erow = tile * 16 + l;
    int eg   = erow < E ? erow : (E - 1);       // clamp: keep EXEC full for WMMA
    const unsigned short* arow0 = fdrug + (long long)src[eg] * 128;
    const unsigned short* arow1 = fdis  + (long long)dst[eg] * 128;
    __builtin_prefetch(arow0, 0, 0);
    __builtin_prefetch(arow1, 0, 0);

    // =================== layer 1: [16x256] x [256x128] ===================
    v8f acc[8] = {};
    #pragma unroll
    for (int kk = 0; kk < 8; ++kk) {
      // A fragment (16x32 bf16): half picks K offsets 0-15/16-31
      const unsigned short* base =
          (kk < 4) ? (arow0 + 32 * kk) : (arow1 + 32 * kk - 128);
      const unsigned short* c1 = base + (hi ? 16 : 0);
      union { uint4 q[2]; v16bf v; } Au;
      Au.q[0] = *(const uint4*)(c1);        // K offsets +0..7
      Au.q[1] = *(const uint4*)(c1 + 8);    // K offsets +8..15

      const int kb = 32 * kk + (hi ? 16 : 0);   // contiguous 16 K in W1^T
      #pragma unroll
      for (int nn = 0; nn < 8; ++nn) {
        v16bf B = __builtin_bit_cast(
            v16bf, *(const v8u*)(w1t + (16 * nn + l) * D_IN + kb));
        acc[nn] = __builtin_amdgcn_wmma_f32_16x16x32_bf16(
            false, Au.v, false, B, (short)0, acc[nn], false, false);
      }
    }

    // ---- bias + LayerNorm + GELU, in registers on the C layout ----
    float cb[8], cg[8], cbe[8];
    #pragma unroll
    for (int nn = 0; nn < 8; ++nn) {
      int c = 16 * nn + l;
      cb[nn] = b1[c]; cg[nn] = g1[c]; cbe[nn] = be1[c];
    }
    #pragma unroll
    for (int nn = 0; nn < 8; ++nn)
      #pragma unroll
      for (int r = 0; r < 8; ++r) acc[nn][r] += cb[nn];

    #pragma unroll
    for (int r = 0; r < 8; ++r) {        // row M = r + 8*hi across 16 lanes
      float sr = 0.f, qr = 0.f;
      #pragma unroll
      for (int nn = 0; nn < 8; ++nn) { float x = acc[nn][r]; sr += x; qr += x * x; }
      sr = hredux16(sr); qr = hredux16(qr);
      float mu   = sr * (1.0f / 128.0f);
      float var  = qr * (1.0f / 128.0f) - mu * mu;
      float rstd = rsqrtf(var + 1e-5f);
      #pragma unroll
      for (int nn = 0; nn < 8; ++nn) {
        float y = (acc[nn][r] - mu) * rstd * cg[nn] + cbe[nn];
        y = gelu_exact(y);
        h1w[(r + 8 * hi) * H1 + 16 * nn + l] = bfbits(y);   // restage for L2
      }
    }

    // =================== layer 2: [16x128] x [128x64] ===================
    v8f acc2[4] = {};
    #pragma unroll
    for (int kk = 0; kk < 4; ++kk) {
      const int o = 32 * kk + (hi ? 16 : 0);
      union { uint4 q[2]; v16bf v; } Au;
      Au.q[0] = *(const uint4*)(h1w + l * H1 + o);
      Au.q[1] = *(const uint4*)(h1w + l * H1 + o + 8);
      const int kb = 32 * kk + (hi ? 16 : 0);
      #pragma unroll
      for (int nn = 0; nn < 4; ++nn) {
        v16bf B = __builtin_bit_cast(
            v16bf, *(const v8u*)(w2t + (16 * nn + l) * H1 + kb));
        acc2[nn] = __builtin_amdgcn_wmma_f32_16x16x32_bf16(
            false, Au.v, false, B, (short)0, acc2[nn], false, false);
      }
    }

    // ---- bias + LN + GELU + final dot with W3 ----
    float cb2[4], cg2[4], cbe2[4], cw3[4];
    #pragma unroll
    for (int nn = 0; nn < 4; ++nn) {
      int c = 16 * nn + l;
      cb2[nn] = b2[c]; cg2[nn] = g2[c]; cbe2[nn] = be2[c]; cw3[nn] = W3[c];
    }
    #pragma unroll
    for (int nn = 0; nn < 4; ++nn)
      #pragma unroll
      for (int r = 0; r < 8; ++r) acc2[nn][r] += cb2[nn];

    float dotv[8];
    #pragma unroll
    for (int r = 0; r < 8; ++r) {
      float sr = 0.f, qr = 0.f;
      #pragma unroll
      for (int nn = 0; nn < 4; ++nn) { float x = acc2[nn][r]; sr += x; qr += x * x; }
      sr = hredux16(sr); qr = hredux16(qr);
      float mu   = sr * (1.0f / 64.0f);
      float var  = qr * (1.0f / 64.0f) - mu * mu;
      float rstd = rsqrtf(var + 1e-5f);
      float p = 0.f;
      #pragma unroll
      for (int nn = 0; nn < 4; ++nn) {
        float y = (acc2[nn][r] - mu) * rstd * cg2[nn] + cbe2[nn];
        p += gelu_exact(y) * cw3[nn];
      }
      dotv[r] = hredux16(p);
    }

    const float bb3 = b3p[0];
    if (l == 0) {                       // lanes 0 and 16 write 8 rows each
      #pragma unroll
      for (int r = 0; r < 8; ++r) {
        int e = tile * 16 + r + 8 * hi;
        if (e < E) out[e] = dotv[r] + bb3;
      }
    }
  }
}

// =====================================================================
// FALLBACK main kernel (self-contained, fp32 gathers + in-block staging)
// — used only if ws_size is too small for the bf16 workspace.
// =====================================================================
__global__ __launch_bounds__(256, 1)
void mlp_edge_decoder_fallback(
    const float* __restrict__ drug, const float* __restrict__ dis,
    const int*   __restrict__ src,  const int*   __restrict__ dst,
    const float* __restrict__ W1, const float* __restrict__ b1,
    const float* __restrict__ g1, const float* __restrict__ be1,
    const float* __restrict__ W2, const float* __restrict__ b2,
    const float* __restrict__ g2, const float* __restrict__ be2,
    const float* __restrict__ W3, const float* __restrict__ b3p,
    float* __restrict__ out, int E)
{
  extern __shared__ __align__(128) unsigned short smem[];
  unsigned short* w1t = smem;
  unsigned short* w2t = smem + LDS_W1T;
  unsigned short* h1s = smem + LDS_W1T + LDS_W2T;

  const int tid = threadIdx.x;
  {
    const int k = tid;
    const float4* wrow = (const float4*)(W1 + k * H1);
    for (int n4 = 0; n4 < H1 / 4; ++n4) {
      float4 v = wrow[n4];
      int n = n4 * 4;
      w1t[(n + 0) * D_IN + k] = bfbits(v.x);
      w1t[(n + 1) * D_IN + k] = bfbits(v.y);
      w1t[(n + 2) * D_IN + k] = bfbits(v.z);
      w1t[(n + 3) * D_IN + k] = bfbits(v.w);
    }
  }
  if (tid < H1) {
    const int k = tid;
    const float4* wrow = (const float4*)(W2 + k * H2);
    for (int n4 = 0; n4 < H2 / 4; ++n4) {
      float4 v = wrow[n4];
      int n = n4 * 4;
      w2t[(n + 0) * H1 + k] = bfbits(v.x);
      w2t[(n + 1) * H1 + k] = bfbits(v.y);
      w2t[(n + 2) * H1 + k] = bfbits(v.z);
      w2t[(n + 3) * H1 + k] = bfbits(v.w);
    }
  }
  __syncthreads();

  const int wave = tid >> 5;
  const int lane = tid & 31;
  const int l    = lane & 15;
  const int hi   = lane >> 4;
  unsigned short* h1w = h1s + wave * (16 * H1);

  const int n_tiles = (E + 15) >> 4;
  const int tile = blockIdx.x * WAVES + wave;
  if (tile < n_tiles) {
    int erow = tile * 16 + l;
    int eg   = erow < E ? erow : (E - 1);
    const float* arow0 = drug + (long long)src[eg] * 128;
    const float* arow1 = dis  + (long long)dst[eg] * 128;

    v8f acc[8] = {};
    #pragma unroll
    for (int kk = 0; kk < 8; ++kk) {
      const float* base = (kk < 4) ? (arow0 + 32 * kk) : (arow1 + 32 * kk - 128);
      const float* c1 = base + (hi ? 16 : 0);
      float4 p0 = *(const float4*)(c1);
      float4 p1 = *(const float4*)(c1 + 4);
      float4 p2 = *(const float4*)(c1 + 8);
      float4 p3 = *(const float4*)(c1 + 12);
      v16bf A;
      A[0]  = __builtin_bit_cast(__bf16, bfbits(p0.x));
      A[1]  = __builtin_bit_cast(__bf16, bfbits(p0.y));
      A[2]  = __builtin_bit_cast(__bf16, bfbits(p0.z));
      A[3]  = __builtin_bit_cast(__bf16, bfbits(p0.w));
      A[4]  = __builtin_bit_cast(__bf16, bfbits(p1.x));
      A[5]  = __builtin_bit_cast(__bf16, bfbits(p1.y));
      A[6]  = __builtin_bit_cast(__bf16, bfbits(p1.z));
      A[7]  = __builtin_bit_cast(__bf16, bfbits(p1.w));
      A[8]  = __builtin_bit_cast(__bf16, bfbits(p2.x));
      A[9]  = __builtin_bit_cast(__bf16, bfbits(p2.y));
      A[10] = __builtin_bit_cast(__bf16, bfbits(p2.z));
      A[11] = __builtin_bit_cast(__bf16, bfbits(p2.w));
      A[12] = __builtin_bit_cast(__bf16, bfbits(p3.x));
      A[13] = __builtin_bit_cast(__bf16, bfbits(p3.y));
      A[14] = __builtin_bit_cast(__bf16, bfbits(p3.z));
      A[15] = __builtin_bit_cast(__bf16, bfbits(p3.w));

      const int kb = 32 * kk + (hi ? 16 : 0);
      #pragma unroll
      for (int nn = 0; nn < 8; ++nn) {
        v16bf B = __builtin_bit_cast(
            v16bf, *(const v8u*)(w1t + (16 * nn + l) * D_IN + kb));
        acc[nn] = __builtin_amdgcn_wmma_f32_16x16x32_bf16(
            false, A, false, B, (short)0, acc[nn], false, false);
      }
    }

    float cb[8], cg[8], cbe[8];
    #pragma unroll
    for (int nn = 0; nn < 8; ++nn) {
      int c = 16 * nn + l;
      cb[nn] = b1[c]; cg[nn] = g1[c]; cbe[nn] = be1[c];
    }
    #pragma unroll
    for (int nn = 0; nn < 8; ++nn)
      #pragma unroll
      for (int r = 0; r < 8; ++r) acc[nn][r] += cb[nn];

    #pragma unroll
    for (int r = 0; r < 8; ++r) {
      float sr = 0.f, qr = 0.f;
      #pragma unroll
      for (int nn = 0; nn < 8; ++nn) { float x = acc[nn][r]; sr += x; qr += x * x; }
      sr = hredux16(sr); qr = hredux16(qr);
      float mu   = sr * (1.0f / 128.0f);
      float var  = qr * (1.0f / 128.0f) - mu * mu;
      float rstd = rsqrtf(var + 1e-5f);
      #pragma unroll
      for (int nn = 0; nn < 8; ++nn) {
        float y = (acc[nn][r] - mu) * rstd * cg[nn] + cbe[nn];
        y = gelu_exact(y);
        h1w[(r + 8 * hi) * H1 + 16 * nn + l] = bfbits(y);
      }
    }

    v8f acc2[4] = {};
    #pragma unroll
    for (int kk = 0; kk < 4; ++kk) {
      const int o = 32 * kk + (hi ? 16 : 0);
      union { uint4 q[2]; v16bf v; } Au;
      Au.q[0] = *(const uint4*)(h1w + l * H1 + o);
      Au.q[1] = *(const uint4*)(h1w + l * H1 + o + 8);
      const int kb = 32 * kk + (hi ? 16 : 0);
      #pragma unroll
      for (int nn = 0; nn < 4; ++nn) {
        v16bf B = __builtin_bit_cast(
            v16bf, *(const v8u*)(w2t + (16 * nn + l) * H1 + kb));
        acc2[nn] = __builtin_amdgcn_wmma_f32_16x16x32_bf16(
            false, Au.v, false, B, (short)0, acc2[nn], false, false);
      }
    }

    float cb2[4], cg2[4], cbe2[4], cw3[4];
    #pragma unroll
    for (int nn = 0; nn < 4; ++nn) {
      int c = 16 * nn + l;
      cb2[nn] = b2[c]; cg2[nn] = g2[c]; cbe2[nn] = be2[c]; cw3[nn] = W3[c];
    }
    #pragma unroll
    for (int nn = 0; nn < 4; ++nn)
      #pragma unroll
      for (int r = 0; r < 8; ++r) acc2[nn][r] += cb2[nn];

    float dotv[8];
    #pragma unroll
    for (int r = 0; r < 8; ++r) {
      float sr = 0.f, qr = 0.f;
      #pragma unroll
      for (int nn = 0; nn < 4; ++nn) { float x = acc2[nn][r]; sr += x; qr += x * x; }
      sr = hredux16(sr); qr = hredux16(qr);
      float mu   = sr * (1.0f / 64.0f);
      float var  = qr * (1.0f / 64.0f) - mu * mu;
      float rstd = rsqrtf(var + 1e-5f);
      float p = 0.f;
      #pragma unroll
      for (int nn = 0; nn < 4; ++nn) {
        float y = (acc2[nn][r] - mu) * rstd * cg2[nn] + cbe2[nn];
        p += gelu_exact(y) * cw3[nn];
      }
      dotv[r] = hredux16(p);
    }

    const float bb3 = b3p[0];
    if (l == 0) {
      #pragma unroll
      for (int r = 0; r < 8; ++r) {
        int e = tile * 16 + r + 8 * hi;
        if (e < E) out[e] = dotv[r] + bb3;
      }
    }
  }
}

// =====================================================================
extern "C" void kernel_launch(void* const* d_in, const int* in_sizes, int n_in,
                              void* d_out, int out_size, void* d_ws, size_t ws_size,
                              hipStream_t stream) {
  (void)n_in; (void)out_size;
  const float* drug = (const float*)d_in[0];
  const float* dis  = (const float*)d_in[1];
  const int*   src  = (const int*)  d_in[2];
  const int*   dst  = (const int*)  d_in[3];
  const float* W1   = (const float*)d_in[4];
  const float* b1   = (const float*)d_in[5];
  const float* g1   = (const float*)d_in[6];
  const float* be1  = (const float*)d_in[7];
  const float* W2   = (const float*)d_in[8];
  const float* b2   = (const float*)d_in[9];
  const float* g2   = (const float*)d_in[10];
  const float* be2  = (const float*)d_in[11];
  const float* W3   = (const float*)d_in[12];
  const float* b3   = (const float*)d_in[13];
  float* out = (float*)d_out;

  const int E = in_sizes[2];
  const int n_tiles = (E + 15) / 16;
  const int blocks  = (n_tiles + WAVES - 1) / WAVES;

  const size_t need =
      ((size_t)in_sizes[0] + (size_t)in_sizes[1] + WPACK_ELEMS) * sizeof(unsigned short);

  if (ws_size >= need) {
    unsigned short* fdrug = (unsigned short*)d_ws;
    unsigned short* fdis  = fdrug + in_sizes[0];
    unsigned short* wpack = fdis + in_sizes[1];

    const int nd4 = in_sizes[0] / 4;
    const int ns4 = in_sizes[1] / 4;
    feat_to_bf16<<<(nd4 + 255) / 256, 256, 0, stream>>>(drug, fdrug, nd4);
    feat_to_bf16<<<(ns4 + 255) / 256, 256, 0, stream>>>(dis, fdis, ns4);
    weights_pack<<<1, 256, 0, stream>>>(W1, W2, wpack);

    (void)hipFuncSetAttribute((const void*)mlp_edge_decoder_fast,
                              hipFuncAttributeMaxDynamicSharedMemorySize, SMEM_BYTES);
    mlp_edge_decoder_fast<<<blocks, 256, SMEM_BYTES, stream>>>(
        fdrug, fdis, wpack, src, dst, b1, g1, be1, b2, g2, be2, W3, b3, out, E);
  } else {
    (void)hipFuncSetAttribute((const void*)mlp_edge_decoder_fallback,
                              hipFuncAttributeMaxDynamicSharedMemorySize, SMEM_BYTES);
    mlp_edge_decoder_fallback<<<blocks, 256, SMEM_BYTES, stream>>>(
        drug, dis, src, dst, W1, b1, g1, be1, W2, b2, g2, be2, W3, b3, out, E);
  }
}
